// MemoryMultiHeadedAttention_11553462026562
// MI455X (gfx1250) — compile-verified, matrix operand order checked by reading, not run on hardware
//
#include <hip/hip_runtime.h>
#include <hip/hip_bf16.h>
#include <math.h>

// ---------------------------------------------------------------------------
// CDNA5 (gfx1250) wave32 WMMA implementation of the compressive-memory
// multi-head attention reference. All matmul operands are pre-packed to bf16
// in memory; every GEMM-like op uses v_wmma_f32_16x16x32_bf16 with fragments
// fetched as two b128 loads per lane (contiguous K runs, ISA 05_wmma 7.12.2).
// GEMM waves compute a 16x64 C strip: A-fragment reused across 4 WMMAs.
// ---------------------------------------------------------------------------

typedef __attribute__((ext_vector_type(16))) __bf16 v16bf;
typedef __attribute__((ext_vector_type(8)))  __bf16 v8bf;
typedef __attribute__((ext_vector_type(8)))  float  v8f;

#define B_   8
#define T_   512
#define D_   512
#define H_   8
#define DH_  64
#define MEMLEN 512
#define CMEMLEN 128
#define KVLEN 1152     // CMEMLEN + MEMLEN + T_
#define NEG_MAX (-3.402823466e38f)

// K index for element e of a 16-bit A/B fragment (ISA 05_wmma 7.12.2)
__device__ __forceinline__ int kmap(int e, int lane) {
    return ((e >> 3) << 4) + (e & 7) + ((lane >> 4) << 3);
}

// Fragment from a bf16 row-major row: lane's elements are two contiguous
// 8-element (16-byte) K runs -> two b128 loads. rowk = &row[k0].
__device__ __forceinline__ v16bf frag_bf16(const __bf16* rowk, int lane) {
    int hi = (lane >> 4) << 3;          // 0 or 8
    v8bf lo  = *(const v8bf*)(rowk + hi);        // K = hi..hi+7
    v8bf hi8 = *(const v8bf*)(rowk + hi + 16);   // K = 16+hi..16+hi+7
    return __builtin_shufflevector(lo, hi8, 0, 1, 2, 3, 4, 5, 6, 7,
                                            8, 9, 10, 11, 12, 13, 14, 15);
}

__device__ __forceinline__ v8f wmma_bf16(v16bf a, v16bf b, v8f c) {
    return __builtin_amdgcn_wmma_f32_16x16x32_bf16(
        false, a, false, b, (short)0, c, false, false);
}

// ---------------------------------------------------------------------------
// Pack / convert kernels
// ---------------------------------------------------------------------------

// 8-wide f32 -> bf16 (n must be a multiple of 8; true for all call sites)
__global__ void cvt8_bf16(const float* __restrict__ in,
                          __bf16* __restrict__ out, long n) {
    long i8 = ((long)blockIdx.x * blockDim.x + threadIdx.x) << 3;
    if (i8 >= n) return;
    float4 a = *(const float4*)(in + i8);
    float4 b = *(const float4*)(in + i8 + 4);
    v8bf o;
    o[0] = (__bf16)a.x; o[1] = (__bf16)a.y; o[2] = (__bf16)a.z; o[3] = (__bf16)a.w;
    o[4] = (__bf16)b.x; o[5] = (__bf16)b.y; o[6] = (__bf16)b.z; o[7] = (__bf16)b.w;
    *(v8bf*)(out + i8) = o;
}

// kv_input = concat([cmem(128), mem(512), x(512)], axis=1) -> bf16 (8,1152,512)
__global__ void pack_kvin(const float* __restrict__ cmem,
                          const float* __restrict__ mem,
                          const float* __restrict__ x,
                          __bf16* __restrict__ kvin) {
    long idx = (long)blockIdx.x * blockDim.x + threadIdx.x;
    const long total = (long)B_ * KVLEN * D_;
    if (idx >= total) return;
    int c = (int)(idx % D_);
    long rr = idx / D_;
    int r = (int)(rr % KVLEN);
    int b = (int)(rr / KVLEN);
    float v;
    if (r < CMEMLEN)             v = cmem[((long)b * CMEMLEN + r) * D_ + c];
    else if (r < CMEMLEN+MEMLEN) v = mem[((long)b * MEMLEN + (r - CMEMLEN)) * D_ + c];
    else                         v = x[((long)b * T_ + (r - CMEMLEN - MEMLEN)) * D_ + c];
    kvin[idx] = (__bf16)v;
}

// im2col (stride-4 width-4 conv): convA[b, ci, ic*4+r] = mem[b, ci*4+r, ic]
__global__ void build_convA(const float* __restrict__ mem, __bf16* __restrict__ convA) {
    long idx = (long)blockIdx.x * blockDim.x + threadIdx.x;
    const long total = (long)B_ * CMEMLEN * 2048;
    if (idx >= total) return;
    int k  = (int)(idx % 2048);
    long t = idx / 2048;
    int ci = (int)(t % CMEMLEN);
    int b  = (int)(t / CMEMLEN);
    int ic = k >> 2, r = k & 3;
    convA[idx] = (__bf16)mem[((long)b * MEMLEN + ci * 4 + r) * D_ + ic];
}

// ---------------------------------------------------------------------------
// Batched C[m,n] = sum_k A[m,k]*B[n,k] (+bias[n]); bf16 A/B, f32 accumulate.
// One wave computes a 16(M) x 64(N) strip: A-frag loaded once per K step,
// 4 independent WMMAs. Optional f32 and/or bf16 outputs.
// grid = (N/64, M/16, batch), block = 32.
// ---------------------------------------------------------------------------
__global__ __launch_bounds__(32)
void gemm_abt_wmma(const __bf16* __restrict__ A, long lda, long abs_,
                   const __bf16* __restrict__ Bm, long ldb,
                   const float* __restrict__ bias,
                   float* __restrict__ Cf, __bf16* __restrict__ Cb,
                   long ldc, long cbs, int K) {
    int lane = threadIdx.x;
    int col  = lane & 15;
    int grp  = lane >> 4;
    long m0 = (long)blockIdx.y << 4;
    long n0 = (long)blockIdx.x << 6;   // 64-wide N strip
    const __bf16* Arow = A + (long)blockIdx.z * abs_ + (m0 + col) * lda;
    const __bf16* B0 = Bm + (n0 +  0 + col) * ldb;
    const __bf16* B1 = Bm + (n0 + 16 + col) * ldb;
    const __bf16* B2 = Bm + (n0 + 32 + col) * ldb;
    const __bf16* B3 = Bm + (n0 + 48 + col) * ldb;

    v8f acc0 = {0.f,0.f,0.f,0.f,0.f,0.f,0.f,0.f};
    v8f acc1 = acc0, acc2 = acc0, acc3 = acc0;
    for (int k0 = 0; k0 < K; k0 += 32) {
        if (k0 + 32 < K) {   // global_prefetch_b8 for the next K tile
            __builtin_prefetch(Arow + k0 + 32, 0, 3);
            __builtin_prefetch(B0 + k0 + 32, 0, 3);
            __builtin_prefetch(B1 + k0 + 32, 0, 3);
            __builtin_prefetch(B2 + k0 + 32, 0, 3);
            __builtin_prefetch(B3 + k0 + 32, 0, 3);
        }
        v16bf af = frag_bf16(Arow + k0, lane);
        acc0 = wmma_bf16(af, frag_bf16(B0 + k0, lane), acc0);
        acc1 = wmma_bf16(af, frag_bf16(B1 + k0, lane), acc1);
        acc2 = wmma_bf16(af, frag_bf16(B2 + k0, lane), acc2);
        acc3 = wmma_bf16(af, frag_bf16(B3 + k0, lane), acc3);
    }
    float bv0 = bias ? bias[n0 +  0 + col] : 0.0f;
    float bv1 = bias ? bias[n0 + 16 + col] : 0.0f;
    float bv2 = bias ? bias[n0 + 32 + col] : 0.0f;
    float bv3 = bias ? bias[n0 + 48 + col] : 0.0f;
    long cb = (long)blockIdx.z * cbs;
#pragma unroll
    for (int r = 0; r < 8; ++r) {
        long m = m0 + r + (grp << 3);
        float v0 = acc0[r] + bv0, v1 = acc1[r] + bv1;
        float v2 = acc2[r] + bv2, v3 = acc3[r] + bv3;
        if (Cf) {
            Cf[cb + m * ldc + n0 +  0 + col] = v0;
            Cf[cb + m * ldc + n0 + 16 + col] = v1;
            Cf[cb + m * ldc + n0 + 32 + col] = v2;
            Cf[cb + m * ldc + n0 + 48 + col] = v3;
        }
        if (Cb) {
            Cb[cb + m * ldc + n0 +  0 + col] = (__bf16)v0;
            Cb[cb + m * ldc + n0 + 16 + col] = (__bf16)v1;
            Cb[cb + m * ldc + n0 + 32 + col] = (__bf16)v2;
            Cb[cb + m * ldc + n0 + 48 + col] = (__bf16)v3;
        }
    }
}

// ---------------------------------------------------------------------------
// Fused attention: one wave = 16 query rows of one (b,h).
// scores(LDS f32) -> softmax -> P(LDS bf16) @ V(LDS-staged bf16), all WMMA.
// causal_off >= 0: mask where j > i + causal_off. All operands bf16.
// grid = (t/16, b*H), block = 32
// ---------------------------------------------------------------------------
__global__ __launch_bounds__(32)
void attn_wmma(const __bf16* __restrict__ q,
               const __bf16* __restrict__ k,
               const __bf16* __restrict__ v,
               long q_bs, long kv_bs, int k_rs, int kv_len, int causal_off,
               float* __restrict__ out, long out_bs) {
    __shared__ float  sc[16][KVLEN + 32];   // raw scores, ~74 KB
    __shared__ __bf16 pb[16][KVLEN + 8];    // normalized probs, ~36 KB (pad=8 -> conflict-free b128)
    __shared__ __bf16 vsm[32][DH_];         // staged V chunk, 4 KB

    int lane = threadIdx.x;
    int col  = lane & 15;
    int grp  = lane >> 4;
    int bh = blockIdx.y;
    int b  = bh >> 3;
    int h  = bh & 7;
    int i0 = blockIdx.x << 4;

    const __bf16* qb = q + (long)b * q_bs  + h * DH_;
    const __bf16* kb = k + (long)b * kv_bs + h * DH_;
    const __bf16* vb = v + (long)b * kv_bs + h * DH_;
    const float scale = 0.125f;  // DH^-0.5

    // Q fragments (16 rows x 64 features -> two K=32 A-frags)
    const __bf16* qrow = qb + (long)(i0 + col) * D_;
    v16bf qa0 = frag_bf16(qrow,      lane);
    v16bf qa1 = frag_bf16(qrow + 32, lane);

    // scores = scale * Q K^T (masked) -> LDS
    for (int j0 = 0; j0 < kv_len; j0 += 16) {
        const __bf16* krow = kb + (long)(j0 + col) * k_rs;
        if (j0 + 16 < kv_len)
            __builtin_prefetch(kb + (long)(j0 + 16 + col) * k_rs, 0, 3);
        v16bf kf0 = frag_bf16(krow,      lane);
        v16bf kf1 = frag_bf16(krow + 32, lane);
        v8f s = {0.f, 0.f, 0.f, 0.f, 0.f, 0.f, 0.f, 0.f};
        s = wmma_bf16(qa0, kf0, s);
        s = wmma_bf16(qa1, kf1, s);
#pragma unroll
        for (int r = 0; r < 8; ++r) {
            int m = r + (grp << 3);
            int j = j0 + col;
            float val = s[r] * scale;
            if (causal_off >= 0 && j > (i0 + m) + causal_off) val = NEG_MAX;
            sc[m][j] = val;
        }
    }
    __syncthreads();

    // row-wise softmax; normalized probs written once as bf16
    for (int m = 0; m < 16; ++m) {
        float mx = NEG_MAX;
        for (int j = lane; j < kv_len; j += 32) mx = fmaxf(mx, sc[m][j]);
#pragma unroll
        for (int o = 16; o; o >>= 1) mx = fmaxf(mx, __shfl_xor(mx, o, 32));
        float sum = 0.f;
        for (int j = lane; j < kv_len; j += 32) {
            float e = __expf(sc[m][j] - mx);
            sc[m][j] = e;
            sum += e;
        }
#pragma unroll
        for (int o = 16; o; o >>= 1) sum += __shfl_xor(sum, o, 32);
        float inv = 1.0f / sum;
        for (int j = lane; j < kv_len; j += 32) pb[m][j] = (__bf16)(sc[m][j] * inv);
    }
    __syncthreads();

    // out = P @ V  (dh = 4 column tiles of 16)
    v8f o0 = {0.f,0.f,0.f,0.f,0.f,0.f,0.f,0.f};
    v8f o1 = o0, o2 = o0, o3 = o0;
    for (int kc = 0; kc < kv_len; kc += 32) {
        // stage V[kc..kc+31][0..63] bf16 -> vsm (256 x 16B, 8 per lane)
#pragma unroll
        for (int i = 0; i < 8; ++i) {
            int idx = lane + (i << 5);        // 0..255
            int row = idx >> 3;               // 8 x 16B per row
            int c8  = idx & 7;
            *(uint4*)&vsm[row][c8 << 3] =
                *(const uint4*)(vb + (long)(kc + row) * k_rs + (c8 << 3));
        }
        __syncthreads();

        v16bf pa = frag_bf16(&pb[col][kc], lane);   // 2 x ds b128
        v16bf vf;
#pragma unroll
        for (int e = 0; e < 16; ++e) vf[e] = vsm[kmap(e, lane)][0 * 16 + col];
        o0 = wmma_bf16(pa, vf, o0);
#pragma unroll
        for (int e = 0; e < 16; ++e) vf[e] = vsm[kmap(e, lane)][1 * 16 + col];
        o1 = wmma_bf16(pa, vf, o1);
#pragma unroll
        for (int e = 0; e < 16; ++e) vf[e] = vsm[kmap(e, lane)][2 * 16 + col];
        o2 = wmma_bf16(pa, vf, o2);
#pragma unroll
        for (int e = 0; e < 16; ++e) vf[e] = vsm[kmap(e, lane)][3 * 16 + col];
        o3 = wmma_bf16(pa, vf, o3);
        __syncthreads();
    }
    float* ob = out + (long)b * out_bs + h * DH_;
#pragma unroll
    for (int r = 0; r < 8; ++r) {
        long m = i0 + r + (grp << 3);
        ob[m * D_ + 0 * 16 + col] = o0[r];
        ob[m * D_ + 1 * 16 + col] = o1[r];
        ob[m * D_ + 2 * 16 + col] = o2[r];
        ob[m * D_ + 3 * 16 + col] = o3[r];
    }
}

// ---------------------------------------------------------------------------
// Reductions
// ---------------------------------------------------------------------------
__global__ void zero_acc(float* acc) { acc[0] = 0.f; acc[1] = 0.f; }

__global__ void sq_diff_reduce(const float* __restrict__ a,
                               const float* __restrict__ b,
                               long n, float* acc) {
    __shared__ float red[256];
    float s = 0.f;
    for (long i = (long)blockIdx.x * blockDim.x + threadIdx.x; i < n;
         i += (long)gridDim.x * blockDim.x) {
        float d = a[i] - b[i];
        s += d * d;
    }
    red[threadIdx.x] = s;
    __syncthreads();
    for (int o = blockDim.x >> 1; o; o >>= 1) {
        if ((int)threadIdx.x < o) red[threadIdx.x] += red[threadIdx.x + o];
        __syncthreads();
    }
    if (threadIdx.x == 0) atomicAdd(acc, red[0]);
}

// recon[b,m,dd] = db[m] + sum_c compressed[b,c,dd]*dw[m,c]; ae += (mem-recon)^2
__global__ void recon_ae(const float* __restrict__ comp,
                         const float* __restrict__ dw,
                         const float* __restrict__ db,
                         const float* __restrict__ mem,
                         float* acc) {
    __shared__ float red[256];
    long idx = (long)blockIdx.x * blockDim.x + threadIdx.x;
    const long total = (long)B_ * MEMLEN * D_;
    float sq = 0.f;
    if (idx < total) {
        int dd = (int)(idx % D_);
        long t = idx / D_;
        int m = (int)(t % MEMLEN);
        int b = (int)(t / MEMLEN);
        float r = db[m];
        const float* cp = comp + (long)b * CMEMLEN * D_ + dd;
        const float* wp = dw + (long)m * CMEMLEN;
        for (int c = 0; c < CMEMLEN; ++c) r += cp[(long)c * D_] * wp[c];
        float d = mem[idx] - r;
        sq = d * d;
    }
    red[threadIdx.x] = sq;
    __syncthreads();
    for (int o = blockDim.x >> 1; o; o >>= 1) {
        if ((int)threadIdx.x < o) red[threadIdx.x] += red[threadIdx.x + o];
        __syncthreads();
    }
    if (threadIdx.x == 0) atomicAdd(acc + 1, red[0]);
}

__global__ void finalize_losses(const float* acc, float* out_aux, float* out_ae) {
    *out_aux = acc[0] / 2097152.0f;  // mean over (8,8,512,64)
    *out_ae  = acc[1] / 2097152.0f;  // mean over (8,512,512)
}

// ---------------------------------------------------------------------------
// Host launch
// ---------------------------------------------------------------------------
extern "C" void kernel_launch(void* const* d_in, const int* in_sizes, int n_in,
                              void* d_out, int out_size, void* d_ws, size_t ws_size,
                              hipStream_t stream) {
    (void)in_sizes; (void)n_in; (void)out_size; (void)ws_size;
    const float* x      = (const float*)d_in[0];
    const float* mem    = (const float*)d_in[1];
    const float* cmem   = (const float*)d_in[2];
    const float* Wq     = (const float*)d_in[3];
    const float* Wkv    = (const float*)d_in[4];
    const float* Wo     = (const float*)d_in[5];
    const float* bo     = (const float*)d_in[6];
    const float* conv_w = (const float*)d_in[7];
    const float* conv_b = (const float*)d_in[8];
    const float* dw     = (const float*)d_in[9];
    const float* db     = (const float*)d_in[10];

    float* out = (float*)d_out;
    float* logits_o = out;                       // 8*512*512
    float* newmem_o = out + 2097152;             // 8*512*512
    float* cmem_o   = out + 4194304;             // 8*128*512 (== compressed)
    float* aux_o    = out + 4718592;
    float* ae_o     = out + 4718593;

    // workspace carve-out (256B aligned chunks)
    char* p = (char*)d_ws;
    auto alloc = [&p](size_t bytes) {
        char* r = p;
        p += (bytes + 255) & ~(size_t)255;
        return r;
    };
    float*  ws_attn  = (float*)alloc(2097152 * 4);
    float*  ws_aux1  = (float*)alloc(2097152 * 4);
    float*  ws_aux2  = (float*)alloc(2097152 * 4);
    float*  ws_acc   = (float*)alloc(2 * 4);
    __bf16* x_bf     = (__bf16*)alloc(2097152 * 2);
    __bf16* kvin_bf  = (__bf16*)alloc(4718592 * 2);
    __bf16* kv_bf    = (__bf16*)alloc(9437184 * 2);
    __bf16* q_bf     = (__bf16*)alloc(2097152 * 2);
    __bf16* attn_bf  = (__bf16*)alloc(2097152 * 2);
    __bf16* convA_bf = (__bf16*)alloc(2097152 * 2);
    __bf16* comp_bf  = (__bf16*)alloc(524288 * 2);
    __bf16* ckv_bf   = (__bf16*)alloc(1048576 * 2);
    __bf16* Wq_bf    = (__bf16*)alloc(262144 * 2);
    __bf16* Wkv_bf   = (__bf16*)alloc(524288 * 2);
    __bf16* Wo_bf    = (__bf16*)alloc(262144 * 2);
    __bf16* convw_bf = (__bf16*)alloc(1048576 * 2);

    zero_acc<<<1, 1, 0, stream>>>(ws_acc);

    // new_mem = x
    hipMemcpyAsync(newmem_o, x, (size_t)2097152 * sizeof(float),
                   hipMemcpyDeviceToDevice, stream);

    // bf16 packing of operands
    cvt8_bf16<<<dim3(1024), 256, 0, stream>>>(x, x_bf, 2097152);
    cvt8_bf16<<<dim3(128),  256, 0, stream>>>(Wq, Wq_bf, 262144);
    cvt8_bf16<<<dim3(256),  256, 0, stream>>>(Wkv, Wkv_bf, 524288);
    cvt8_bf16<<<dim3(128),  256, 0, stream>>>(Wo, Wo_bf, 262144);
    cvt8_bf16<<<dim3(512),  256, 0, stream>>>(conv_w, convw_bf, 1048576);
    pack_kvin<<<dim3(18432), 256, 0, stream>>>(cmem, mem, x, kvin_bf);
    build_convA<<<dim3(8192), 256, 0, stream>>>(mem, convA_bf);

    // q = x @ Wq^T (bf16 out only; consumed as WMMA operand)
    gemm_abt_wmma<<<dim3(8, 32, 8), 32, 0, stream>>>(
        x_bf, 512, 262144, Wq_bf, 512, nullptr,
        nullptr, q_bf, 512, 262144, 512);

    // kv = kv_input @ Wkv^T (bf16 out only)
    gemm_abt_wmma<<<dim3(16, 72, 8), 32, 0, stream>>>(
        kvin_bf, 512, 589824, Wkv_bf, 512, nullptr,
        nullptr, kv_bf, 1024, 1179648, 512);

    // main causal attention over full kv (1152), mask j > i + 640
    attn_wmma<<<dim3(32, 64), 32, 0, stream>>>(
        q_bf, kv_bf, kv_bf + 512, 262144, 1179648, 1024, KVLEN, 640,
        ws_attn, 262144);
    cvt8_bf16<<<dim3(1024), 256, 0, stream>>>(ws_attn, attn_bf, 2097152);

    // logits = attn_out @ Wo^T + bo (f32 out)
    gemm_abt_wmma<<<dim3(8, 32, 8), 32, 0, stream>>>(
        attn_bf, 512, 262144, Wo_bf, 512, bo,
        logits_o, nullptr, 512, 262144, 512);

    // compressed (== new_cmem) = convA @ conv_w^T + conv_b (f32 + bf16 outs)
    gemm_abt_wmma<<<dim3(8, 8, 8), 32, 0, stream>>>(
        convA_bf, 2048, 262144, convw_bf, 2048, conv_b,
        cmem_o, comp_bf, 512, 65536, 2048);

    // ckv = compressed @ Wkv^T (bf16 out only)
    gemm_abt_wmma<<<dim3(16, 8, 8), 32, 0, stream>>>(
        comp_bf, 512, 65536, Wkv_bf, 512, nullptr,
        nullptr, ckv_bf, 1024, 131072, 512);

    // aux attention 1: q vs old-mem keys (kv rows 128..639), no mask
    attn_wmma<<<dim3(32, 64), 32, 0, stream>>>(
        q_bf, kv_bf + (long)128 * 1024, kv_bf + (long)128 * 1024 + 512,
        262144, 1179648, 1024, 512, -1, ws_aux1, 262144);

    // aux attention 2: q vs compressed keys (128), no mask
    attn_wmma<<<dim3(32, 64), 32, 0, stream>>>(
        q_bf, ckv_bf, ckv_bf + 512, 262144, 131072, 1024, 128, -1,
        ws_aux2, 262144);

    // aux_loss = mean((aux1 - aux2)^2)
    sq_diff_reduce<<<dim3(2048), 256, 0, stream>>>(ws_aux1, ws_aux2, 2097152, ws_acc);

    // ae_loss = mean((mem - recon)^2)
    recon_ae<<<dim3(8192), 256, 0, stream>>>(cmem_o, dw, db, mem, ws_acc);

    finalize_losses<<<1, 1, 0, stream>>>(ws_acc, aux_o, ae_o);
}
